// NormalEstimator_46677704573370
// MI455X (gfx1250) — matrix operand order repaired and checked by reading, not compiled, not run on hardware
//
#include <hip/hip_runtime.h>
#include <math.h>

typedef __attribute__((ext_vector_type(16))) _Float16 v16h;
typedef __attribute__((ext_vector_type(8)))  _Float16 v8h;
typedef __attribute__((ext_vector_type(8)))  float    v8f;
typedef __attribute__((ext_vector_type(4)))  unsigned int u32x4;
typedef __attribute__((ext_vector_type(4)))  int i32x4;
typedef __attribute__((ext_vector_type(8)))  int i32x8;

#define NB 4
#define KSEG 8
#define HH 320
#define WW 320
#define HW (HH*WW)          // 102400
#define NP (NB*HW)          // 409600
#define NTILE (NP/16)       // 25600 pixel tiles (16 | 320, tiles never cross rows)

// ---------------- workspace layout (bytes) ----------------
static const size_t OFF_F     = 0;                         // f16 [NP][192] feature cat buffer
static const size_t SZ_F      = (size_t)NP*192*2;          // 157,286,400
static const size_t OFF_T     = OFF_F + SZ_F;              // f16 [NP][64] temp (w1 out / daspp out)
static const size_t SZ_T      = (size_t)NP*64*2;           // 52,428,800
static const size_t OFF_WPREP = OFF_T + SZ_T;              // f16 swizzled A-fragments
static const size_t SZ_WPREP  = 262144;
static const size_t OFF_SB    = OFF_WPREP + SZ_WPREP;      // folded BN params (fp32 + f16)
static const size_t OFF_SBH   = OFF_SB + 6400;             // f16 scale0h[480], bias0h[480]
static const size_t OFF_NRM   = OFF_SB + 16384;            // fp32 normals [N][K][3]

// fragment sub-offsets, in f16 elements inside WPREP
static const int W1FRAG = 0;          // 60 frags  (w1: layers, kc-major then 4 cout tiles)
static const int W2FRAG = 30720;      // 180 frags (w2: 5 layers x 18 chunks x 2 tiles)
static const int DAFRAG = 122880;     // 12 frags  (daspp: 6 kc x 2 tiles)
static const int PLFRAG = 129024;     // 1 frag    (plane, rows >=3 zero-padded)

#define W2ELEMS (36*2*512)            // 36864 f16 = 72KB per conv3x3 layer

// scale/bias sub-offsets in floats inside SB (fp32 part)
#define SC1 0
#define BI1 320
__device__ __host__ inline int bn0_layer_off(int L){ const int o[5]={0,32,96,192,320}; return o[L]; }

__device__ inline v8f wmma_f16(v16h a, v16h b, v8f c) {
  return __builtin_amdgcn_wmma_f32_16x16x32_f16(false, a, false, b, (short)0, c, false, false);
}

// ================= phase 1: least-squares plane fit =================
__global__ void ls_accum_kernel(const float* __restrict__ depth,
                                const float* __restrict__ kinv,
                                const int*   __restrict__ seg,
                                float* __restrict__ normals) {
  int nk = blockIdx.x, n = nk / KSEG;
  const float* dep = depth + (size_t)n*HW;
  const int*   msk = seg + (size_t)nk*HW;
  float a[10] = {0,0,0,0,0,0,0,0,0,0};   // m00 m01 m02 m11 m12 m22 s0 s1 s2 cnt
  for (int p = threadIdx.x; p < HW; p += blockDim.x) {
    if (msk[p]) {
      float d  = dep[p];
      float k0 = kinv[p], k1 = kinv[HW+p], k2 = kinv[2*HW+p];
      float w  = d*d;
      a[0] += k0*k0*w; a[1] += k0*k1*w; a[2] += k0*k2*w;
      a[3] += k1*k1*w; a[4] += k1*k2*w; a[5] += k2*k2*w;
      a[6] += k0*d;    a[7] += k1*d;    a[8] += k2*d;    a[9] += 1.f;
    }
  }
  #pragma unroll
  for (int off = 16; off; off >>= 1)
    #pragma unroll
    for (int t = 0; t < 10; ++t) a[t] += __shfl_down(a[t], off);
  __shared__ float red[8][10];
  int wv = threadIdx.x >> 5, ln = threadIdx.x & 31;
  if (ln == 0) for (int t = 0; t < 10; ++t) red[wv][t] = a[t];
  __syncthreads();
  if (threadIdx.x == 0) {
    float r[10];
    for (int t = 0; t < 10; ++t) { r[t] = red[0][t]; for (int w2 = 1; w2 < 8; ++w2) r[t] += red[w2][t]; }
    float* out = normals + (size_t)nk*3;
    if (r[9] <= 0.f) { out[0]=out[1]=out[2]=0.f; return; }
    float m00=r[0],m01=r[1],m02=r[2],m11=r[3],m12=r[4],m22=r[5];
    float c00 =  (m11*m22 - m12*m12);
    float c01 = -(m01*m22 - m12*m02);
    float c02 =  (m01*m12 - m11*m02);
    float det = m00*c00 + m01*c01 + m02*c02;
    float inv = (det != 0.f) ? 1.f/det : 0.f;
    float i00=c00*inv, i01=c01*inv, i02=c02*inv;
    float i11=(m00*m22-m02*m02)*inv, i12=-(m00*m12-m01*m02)*inv, i22=(m00*m11-m01*m01)*inv;
    float v0 = i00*r[6]+i01*r[7]+i02*r[8];
    float v1 = i01*r[6]+i11*r[7]+i12*r[8];
    float v2 = i02*r[6]+i12*r[7]+i22*r[8];
    float nr = sqrtf(v0*v0+v1*v1+v2*v2);
    float sf = (nr > 0.f) ? 1.f/nr : 1.f;
    out[0]=v0*sf; out[1]=v1*sf; out[2]=v2*sf;
  }
}

__global__ void scatter_rough_kernel(const int* __restrict__ seg,
                                     const float* __restrict__ normals,
                                     float* __restrict__ rough) {
  __shared__ float snm[KSEG*3];
  int idx = blockIdx.x*blockDim.x + threadIdx.x;   // over N*HW, blocks never cross images
  int n = idx / HW, p = idx % HW;
  if (threadIdx.x < KSEG*3) snm[threadIdx.x] = normals[(size_t)n*KSEG*3 + threadIdx.x];
  __syncthreads();
  float a0=0.f, a1=0.f, a2=0.f;
  #pragma unroll
  for (int k = 0; k < KSEG; ++k) {
    if (seg[((size_t)(n*KSEG+k))*HW + p]) {
      a0 += snm[k*3+0]; a1 += snm[k*3+1]; a2 += snm[k*3+2];
    }
  }
  rough[((size_t)n*3+0)*HW + p] = a0;
  rough[((size_t)n*3+1)*HW + p] = a1;
  rough[((size_t)n*3+2)*HW + p] = a2;
}

// ================= weight prep: BN folding + A-fragment swizzle =================
struct PrepArgs {
  const float* w1[5];
  const float* w2[5];
  const float* bn0[4][4];   // g,b,m,v for d6,d12,d18,d24
  const float* bn1[5][4];
  const float* daspp;
  const float* plane;
};

__global__ void prep_kernel(PrepArgs A, _Float16* __restrict__ wprep,
                            float* __restrict__ sb, _Float16* __restrict__ sbh) {
  int f = blockIdx.x;
  if (f == 253) {                                 // BN fold section
    for (int i = threadIdx.x; i < 800; i += blockDim.x) {
      if (i < 480) {
        const int offs[6] = {0,32,96,192,320,480};
        int L = 0; while (i >= offs[L+1]) ++L;
        int ch = i - offs[L];
        float s = 1.f, b = 0.f;
        if (L > 0) {
          const float* g = A.bn0[L-1][0]; const float* bb = A.bn0[L-1][1];
          const float* m = A.bn0[L-1][2]; const float* v  = A.bn0[L-1][3];
          s = g[ch] * rsqrtf(v[ch] + 1.1e-5f);
          b = bb[ch] - m[ch]*s;
        }
        sbh[i] = (_Float16)s;            // scale0h
        sbh[480 + i] = (_Float16)b;      // bias0h
      } else {
        int j = i - 480, L = j >> 6, ch = j & 63;
        const float* g = A.bn1[L][0]; const float* bb = A.bn1[L][1];
        const float* m = A.bn1[L][2]; const float* v  = A.bn1[L][3];
        float s = g[ch] * rsqrtf(v[ch] + 1e-5f);
        sb[SC1+j] = s; sb[BI1+j] = bb[ch] - m[ch]*s;
      }
    }
    return;
  }
  for (int e = threadIdx.x; e < 512; e += blockDim.x) {
    int lane = e >> 4, j = e & 15;
    int m = lane & 15, half = lane >> 4;
    int kk = half*16 + j;   // contiguous k order (matches B fills)
    float val = 0.f; size_t dst = 0;
    if (f < 60) {                                // w1 fragments
      const int cum[6] = {0,4,12,24,40,60};
      int L = 0; while (f >= cum[L+1]) ++L;
      int within = f - cum[L];
      int kc = within >> 2, tile = within & 3;
      int cin = 32*(L+1);
      val = A.w1[L][(tile*16+m)*cin + kc*32 + kk];
      dst = W1FRAG + (size_t)f*512 + e;
    } else if (f < 240) {                        // w2 fragments (tap-major K)
      int rel = f - 60, L = rel/36, w = rel%36;
      int chunk = w >> 1, tile = w & 1;
      int tap = chunk >> 1, cihalf = chunk & 1;
      val = A.w2[L][((tile*16+m)*64 + cihalf*32 + kk)*9 + tap];
      dst = W2FRAG + (size_t)rel*512 + e;
    } else if (f < 252) {                        // daspp fragments
      int rel = f - 240, kc = rel >> 1, tile = rel & 1;
      val = A.daspp[(tile*16+m)*192 + kc*32 + kk];
      dst = DAFRAG + (size_t)rel*512 + e;
    } else {                                     // plane fragment (rows 3..15 zero)
      val = (m < 3) ? A.plane[m*32 + kk] : 0.f;
      dst = PLFRAG + (size_t)e;
    }
    wprep[dst] = (_Float16)val;
  }
}

// ================= conv0: 3->32, 3x3, pad 1, +bias (direct, tiny) =================
__global__ void conv0_kernel(const float* __restrict__ rough,
                             const float* __restrict__ w,
                             const float* __restrict__ bias,
                             _Float16* __restrict__ F) {
  __shared__ float sw[896];
  for (int i = threadIdx.x; i < 896; i += blockDim.x)
    sw[i] = (i < 864) ? w[i] : bias[i-864];
  __syncthreads();
  int idx = blockIdx.x*blockDim.x + threadIdx.x;      // over N*HW (exact multiple)
  int n = idx / HW, rem = idx % HW, y = rem / WW, x = rem % WW;
  float patch[27];
  #pragma unroll
  for (int ci = 0; ci < 3; ++ci)
    #pragma unroll
    for (int ky = 0; ky < 3; ++ky)
      #pragma unroll
      for (int kx = 0; kx < 3; ++kx) {
        int sy = y + ky - 1, sx = x + kx - 1;
        patch[ci*9+ky*3+kx] = (sy>=0 && sy<HH && sx>=0 && sx<WW)
            ? rough[((size_t)n*3+ci)*HW + sy*WW + sx] : 0.f;
      }
  _Float16* dst = F + (size_t)idx*192;
  for (int co = 0; co < 32; ++co) {
    float a = sw[864+co];
    const float* wc = sw + co*27;
    #pragma unroll
    for (int t = 0; t < 27; ++t) a = fmaf(wc[t], patch[t], a);
    dst[co] = (_Float16)a;
  }
}

// ================= 1x1 conv as WMMA GEMM =================
// IN_OP: 0 = raw, 1 = bn0 fold + relu on gathered activations (packed f16 math)
// OUT_OP: 0 = bn1+relu -> f16, 1 = elu -> f16, 2 = fp32 scatter to d_out (plane)
template<int IN_OP, int OUT_OP>
__global__ void gemm1x1_kernel(const _Float16* __restrict__ in, int in_stride, int cin,
                               const _Float16* __restrict__ frags, int cout_tiles,
                               const _Float16* __restrict__ sc0h, const _Float16* __restrict__ bi0h,
                               _Float16* __restrict__ outh, int out_stride,
                               const float* __restrict__ sc1, const float* __restrict__ bi1,
                               float* __restrict__ outf) {
  int wv = blockIdx.x*(blockDim.x>>5) + (threadIdx.x>>5);
  wv = __builtin_amdgcn_readfirstlane(wv);             // wave-uniform -> scalar branches
  if (wv >= NTILE*cout_tiles) return;
  int ptile = wv % NTILE, ctile = wv / NTILE;
  int lane = threadIdx.x & 31, col = lane & 15, half = lane >> 4;
  int pixel = ptile*16 + col;
  const _Float16* brow = in + (size_t)pixel*in_stride + half*16;
  int kchunks = cin >> 5;
  v8f accA = {}, accB = {};                            // 2 accs: hide WMMA RAW latency
  int kc = 0;
  for (; kc + 2 <= kchunks; kc += 2) {
    v16h a0 = *(const v16h*)(frags + ((size_t)((kc+0)*cout_tiles + ctile))*512 + lane*16);
    v16h b0 = *(const v16h*)(brow + (kc+0)*32);
    v16h a1 = *(const v16h*)(frags + ((size_t)((kc+1)*cout_tiles + ctile))*512 + lane*16);
    v16h b1 = *(const v16h*)(brow + (kc+1)*32);
    if (IN_OP == 1) {
      v16h zero = {};
      int cb0 = (kc+0)*32 + half*16, cb1 = (kc+1)*32 + half*16;
      b0 = __builtin_elementwise_max(b0 * *(const v16h*)(sc0h+cb0) + *(const v16h*)(bi0h+cb0), zero);
      b1 = __builtin_elementwise_max(b1 * *(const v16h*)(sc0h+cb1) + *(const v16h*)(bi0h+cb1), zero);
    }
    accA = wmma_f16(a0, b0, accA);
    accB = wmma_f16(a1, b1, accB);
  }
  if (kc < kchunks) {
    v16h a = *(const v16h*)(frags + ((size_t)(kc*cout_tiles + ctile))*512 + lane*16);
    v16h b = *(const v16h*)(brow + kc*32);
    if (IN_OP == 1) {
      v16h zero = {};
      int cb = kc*32 + half*16;
      b = __builtin_elementwise_max(b * *(const v16h*)(sc0h+cb) + *(const v16h*)(bi0h+cb), zero);
    }
    accA = wmma_f16(a, b, accA);
  }
  v8f acc = accA + accB;
  if (OUT_OP == 0) {
    int cb = ctile*16 + half*8;
    v8h o;
    #pragma unroll
    for (int r = 0; r < 8; ++r)
      o[r] = (_Float16)fmaxf(acc[r]*sc1[cb+r] + bi1[cb+r], 0.f);
    *(v8h*)(outh + (size_t)pixel*out_stride + cb) = o;
  } else if (OUT_OP == 1) {
    int cb = ctile*16 + half*8;
    v8h o;
    #pragma unroll
    for (int r = 0; r < 8; ++r) {
      float g = acc[r];
      o[r] = (_Float16)(g > 0.f ? g : (expf(g) - 1.f));
    }
    *(v8h*)(outh + (size_t)pixel*out_stride + cb) = o;
  } else {
    int n = pixel / HW, rem = pixel % HW;
    #pragma unroll
    for (int r = 0; r < 8; ++r) {
      int m = half*8 + r;
      if (m < 3) outf[((size_t)n*3 + m)*HW + rem] = acc[r];
    }
  }
}

// ================= dilated 3x3 conv 64->32 as implicit-GEMM WMMA =================
// Weight fragments (72KB) staged once per workgroup into LDS via the
// Tensor Data Mover; waves read A-fragments with ds_load_b128.
__global__ void conv3x3_kernel(const _Float16* __restrict__ T,
                               const _Float16* __restrict__ frags,
                               _Float16* __restrict__ F, int coff, int dil) {
  __shared__ _Float16 sA[W2ELEMS];                     // 36 chunks x 2 tiles x 512
#if __has_builtin(__builtin_amdgcn_tensor_load_to_lds)
  if (threadIdx.x < 32) {                              // wave 0 issues one TDM DMA
    unsigned lds_base = (unsigned)(uintptr_t)sA;       // flat LDS addr low 32b = LDS offset
    unsigned long long ga = (unsigned long long)(uintptr_t)frags;
    // D# group 0: count=1 | lds_addr | global_addr[56:0] | type=2
    u32x4 g0 = { 1u, lds_base,
                 (unsigned)(ga & 0xffffffffu),
                 (unsigned)((ga >> 32) & 0x1ffffffu) | (2u << 30) };
    // D# group 1: data_size=2B, 1-D transfer of W2ELEMS halfwords
    i32x8 g1 = { (int)0x00010000u,                     // data_size=1 (2 bytes)
                 (int)((W2ELEMS & 0xffff) << 16),      // tensor_dim0[15:0]
                 (int)(((W2ELEMS >> 16) & 0xffff) | (1 << 16)),  // dim0 hi | tensor_dim1=1
                 (int)((W2ELEMS & 0xffff) << 16),      // tile_dim0 = W2ELEMS
                 1,                                    // tile_dim1 = 1
                 (int)W2ELEMS,                         // tensor_dim0_stride lo
                 (int)((W2ELEMS & 0xffff) << 16),      // stride0 hi=0 | tensor_dim1_stride lo
                 0 };
    i32x4 z4 = {0,0,0,0};
    i32x8 z8 = {0,0,0,0,0,0,0,0};
    __builtin_amdgcn_tensor_load_to_lds(g0, g1, z4, z4, z8, 0);
    __builtin_amdgcn_s_wait_tensorcnt(0);
  }
  __syncthreads();
#else
  for (int i = threadIdx.x; i < W2ELEMS/8; i += blockDim.x)
    ((v8h*)sA)[i] = ((const v8h*)frags)[i];
  __syncthreads();
#endif
  int wv = blockIdx.x*(blockDim.x>>5) + (threadIdx.x>>5);
  wv = __builtin_amdgcn_readfirstlane(wv);             // tile coords become SGPRs
  if (wv >= NTILE) return;
  int lane = threadIdx.x & 31, col = lane & 15, half = lane >> 4;
  int p0 = wv*16;
  int n = p0 / HW, rem = p0 % HW, y = rem / WW, x0 = rem % WW;
  int x = x0 + col;
  bool interior = (x0 >= dil) && (x0 + 15 + dil < WW);  // scalar: no per-lane masking
  __builtin_prefetch(T + (size_t)p0*64, 0, 0);          // global_prefetch_b8
  v8f acc00 = {}, acc01 = {}, acc10 = {}, acc11 = {};   // [cihalf][tile] accumulators
  #pragma unroll
  for (int ky = 0; ky < 3; ++ky) {
    int sy = y + (ky-1)*dil;
    if (sy < 0 || sy >= HH) continue;                   // scalar skip: zero contribution
    const _Float16* rowp = T + ((size_t)n*HW + (size_t)sy*WW)*64 + half*16;
    #pragma unroll
    for (int kx = 0; kx < 3; ++kx) {
      int tap = ky*3 + kx;
      int sx = x + (kx-1)*dil;
      const _Float16* src = rowp + (size_t)sx*64;
      bool colok = (sx >= 0) && (sx < WW);
      v16h b0, b1;
      if (interior) {
        b0 = *(const v16h*)(src);
        b1 = *(const v16h*)(src + 32);
      } else {
        b0 = v16h{}; b1 = v16h{};
        if (colok) { b0 = *(const v16h*)(src); b1 = *(const v16h*)(src + 32); }
      }
      const _Float16* fr = sA + (size_t)(tap*2)*2*512 + lane*16;
      v16h a00 = *(const v16h*)(fr + 0*512);            // chunk=tap*2+0, tile 0
      v16h a01 = *(const v16h*)(fr + 1*512);            // chunk=tap*2+0, tile 1
      v16h a10 = *(const v16h*)(fr + 2*512);            // chunk=tap*2+1, tile 0
      v16h a11 = *(const v16h*)(fr + 3*512);            // chunk=tap*2+1, tile 1
      acc00 = wmma_f16(a00, b0, acc00);
      acc01 = wmma_f16(a01, b0, acc01);
      acc10 = wmma_f16(a10, b1, acc10);
      acc11 = wmma_f16(a11, b1, acc11);
    }
  }
  v8f f0 = acc00 + acc10;                               // cout tile 0 (ci 0-31 + 32-63)
  v8f f1 = acc01 + acc11;                               // cout tile 1
  _Float16* dst = F + (size_t)(p0+col)*192 + coff + half*8;
  v8h o0, o1;
  #pragma unroll
  for (int r = 0; r < 8; ++r) { o0[r] = (_Float16)f0[r]; o1[r] = (_Float16)f1[r]; }
  *(v8h*)(dst)      = o0;   // channels coff + half*8 + r
  *(v8h*)(dst + 16) = o1;   // channels coff + 16 + half*8 + r
}

// ================= host launcher =================
extern "C" void kernel_launch(void* const* d_in, const int* in_sizes, int n_in,
                              void* d_out, int out_size, void* d_ws, size_t ws_size,
                              hipStream_t stream) {
  const float* depth = (const float*)d_in[0];
  const float* kinv  = (const float*)d_in[1];
  const int*   seg   = (const int*)d_in[2];

  char* ws = (char*)d_ws;
  _Float16* Fbuf  = (_Float16*)(ws + OFF_F);
  _Float16* Tbuf  = (_Float16*)(ws + OFF_T);
  _Float16* wprep = (_Float16*)(ws + OFF_WPREP);
  float*    sb    = (float*)(ws + OFF_SB);
  _Float16* sbh   = (_Float16*)(ws + OFF_SBH);
  float*    nrm   = (float*)(ws + OFF_NRM);
  float*    rough  = (float*)d_out;
  float*    refine = (float*)d_out + (size_t)NB*3*HW;

  // param index map (flattened setup_inputs order)
  PrepArgs A;
  const int w1i[5]  = {5, 15, 25, 35, 45};
  const int w2i[5]  = {10, 20, 30, 40, 50};
  const int bn0i[4] = {11, 21, 31, 41};
  const int bn1i[5] = {6, 16, 26, 36, 46};
  for (int L = 0; L < 5; ++L) {
    A.w1[L] = (const float*)d_in[w1i[L]];
    A.w2[L] = (const float*)d_in[w2i[L]];
    for (int t = 0; t < 4; ++t) A.bn1[L][t] = (const float*)d_in[bn1i[L]+t];
  }
  for (int L = 0; L < 4; ++L)
    for (int t = 0; t < 4; ++t) A.bn0[L][t] = (const float*)d_in[bn0i[L]+t];
  A.daspp = (const float*)d_in[51];
  A.plane = (const float*)d_in[52];

  // 1) weight prep (BN fold + WMMA A-fragment swizzle)
  prep_kernel<<<254, 256, 0, stream>>>(A, wprep, sb, sbh);

  // 2) least-squares plane fit + scatter rough normals
  ls_accum_kernel<<<NB*KSEG, 256, 0, stream>>>(depth, kinv, seg, nrm);
  scatter_rough_kernel<<<NP/256, 256, 0, stream>>>(seg, nrm, rough);

  // 3) conv0 -> F channels [0,32)
  conv0_kernel<<<NP/256, 256, 0, stream>>>(rough, (const float*)d_in[3],
                                           (const float*)d_in[4], Fbuf);

  // 4) five atrous blocks
  const int cum[5] = {0, 4, 12, 24, 40};   // w1 fragment bases (in frags)
  const int dils[5] = {3, 6, 12, 18, 24};
  for (int L = 0; L < 5; ++L) {
    int cin = 32*(L+1);
    gemm1x1_kernel<1,0><<<(NTILE*4)/8, 256, 0, stream>>>(
        Fbuf, 192, cin, wprep + W1FRAG + (size_t)cum[L]*512, 4,
        sbh + bn0_layer_off(L), sbh + 480 + bn0_layer_off(L),
        Tbuf, 64, sb + SC1 + L*64, sb + BI1 + L*64, nullptr);
    conv3x3_kernel<<<NTILE/8, 256, 0, stream>>>(
        Tbuf, wprep + W2FRAG + (size_t)L*36*512, Fbuf, 32 + L*32, dils[L]);
  }

  // 5) daspp 192->32 + ELU (into Tbuf, stride 64, channels [0,32))
  gemm1x1_kernel<0,1><<<(NTILE*2)/8, 256, 0, stream>>>(
      Fbuf, 192, 192, wprep + DAFRAG, 2,
      nullptr, nullptr, Tbuf, 64, nullptr, nullptr, nullptr);

  // 6) plane 32->3 -> refine (fp32, NCHW)
  gemm1x1_kernel<0,2><<<NTILE/8, 256, 0, stream>>>(
      Tbuf, 64, 32, wprep + PLFRAG, 1,
      nullptr, nullptr, nullptr, 0, nullptr, nullptr, refine);
}
